// Model_58703613001871
// MI455X (gfx1250) — compile-verified, matrix operand order checked by reading, not run on hardware
//
#include <hip/hip_runtime.h>
#include <math.h>

typedef __attribute__((ext_vector_type(16))) __bf16 v16bf;
typedef __attribute__((ext_vector_type(8)))  float  v8f;

#define NBn 65536
#define En  1048576
#define Bn  128
#define Nn  512
#define Fn  128
#define K1n 410
#define K2n 205
#define K3n 103

// ---------------- wave helpers (wave32) ----------------
__device__ __forceinline__ float wred32(float v){
#pragma unroll
  for (int o = 16; o > 0; o >>= 1) v += __shfl_xor(v, o, 32);
  return v;
}
__device__ __forceinline__ float hred8(float v){
  v += __shfl_xor(v, 1, 32);
  v += __shfl_xor(v, 2, 32);
  v += __shfl_xor(v, 4, 32);
  return v;
}
__device__ __forceinline__ unsigned f2o(float f){
  unsigned u = __float_as_uint(f);
  return (u & 0x80000000u) ? ~u : (u | 0x80000000u);
}
__device__ __forceinline__ float o2f(unsigned u){
  unsigned v = (u & 0x80000000u) ? (u & 0x7fffffffu) : ~u;
  return __uint_as_float(v);
}

// ---------------- WMMA bf16 GEMM (16x16x32, f32 acc), K = 128 ----------------
// Fragment layout (16-bit A 16x32): lanes 0-15 -> M=lane, K groups {0..7,16..23};
// lanes 16-31 -> M=lane-16, K groups {8..15,24..31}; element j of v16bf is
// klocal = (j&7) + (j>=8 ? 16 : 0) (+8 for high lanes).

// Pack W[128 x N] (fp32 row-major) into fragment-major bf16 so each lane's
// 16 B values are one contiguous 32-byte load.
__global__ void k_packw(const float* __restrict__ W, __bf16* __restrict__ out, int N){
  int t = blockIdx.x * blockDim.x + threadIdx.x;   // one thread per (frag,lane,v) pair
  int total = (N >> 4) * 4 * 32 * 8;
  if (t >= total) return;
  int v    = t & 7;
  int lane = (t >> 3) & 31;
  int fb   = t >> 8;            // fb = ct*4 + kb
  int kb   = fb & 3;
  int ct   = fb >> 2;
  int kp = (v < 4) ? (2 * v) : (16 + 2 * (v - 4));
  int k  = kb * 32 + ((lane & 16) ? 8 : 0) + kp;
  int n  = ct * 16 + (lane & 15);
  size_t o = ((size_t)fb * 32 + lane) * 16 + 2 * v;
  out[o]     = (__bf16)W[(size_t)k * N + n];
  out[o + 1] = (__bf16)W[(size_t)(k + 1) * N + n];
}

// out = act( A@W [+ A2@W2] + bias ).  One block per 16-row stripe; one wave per
// 16-col tile.  A stripe staged once per block in LDS as bf16 fragments.
template <bool DUAL>
__global__ void k_gemm(const float* __restrict__ A, const __bf16* __restrict__ Wp,
                       const float* __restrict__ A2, const __bf16* __restrict__ Wp2,
                       const float* __restrict__ bias, float* __restrict__ out,
                       int N, int act){
  __shared__ __align__(32) __bf16 la[DUAL ? 2 : 1][4 * 32 * 16];   // 4 kb-steps
  const int row0 = blockIdx.x << 4;
  // ---- cooperative stage: 16 x 128 fp32 -> bf16 fragment layout ----
  for (int e = threadIdx.x; e < 16 * 128; e += blockDim.x) {
    int m = e >> 7, k = e & 127;
    int kb = k >> 5, kl = k & 31;
    int lane = m + ((kl & 8) ? 16 : 0);
    int j    = (kl & 7) + ((kl & 16) ? 8 : 0);
    int slot = (kb * 32 + lane) * 16 + j;
    la[0][slot] = (__bf16)A[(size_t)(row0 + m) * 128 + k];
    if (DUAL) la[1][slot] = (__bf16)A2[(size_t)(row0 + m) * 128 + k];
  }
  __syncthreads();

  const int lane = threadIdx.x & 31;
  const int wave = threadIdx.x >> 5;         // = column tile
  const int col0 = wave << 4;
  const v16bf* ap  = (const v16bf*)la[0] + lane;
  const v16bf* ap2 = (const v16bf*)la[DUAL ? 1 : 0] + lane;
  const v16bf* bp  = (const v16bf*)Wp  + (size_t)wave * 4 * 32 + lane;
  const v16bf* bp2 = DUAL ? ((const v16bf*)Wp2 + (size_t)wave * 4 * 32 + lane) : bp;
  v8f acc = {};
#pragma unroll
  for (int kb = 0; kb < 4; ++kb) {
    v16bf a = ap[kb * 32];
    v16bf b = bp[kb * 32];
    acc = __builtin_amdgcn_wmma_f32_16x16x32_bf16(false, a, false, b, (short)0, acc, false, false);
    if (DUAL) {
      v16bf a2 = ap2[kb * 32];
      v16bf b2 = bp2[kb * 32];
      acc = __builtin_amdgcn_wmma_f32_16x16x32_bf16(false, a2, false, b2, (short)0, acc, false, false);
    }
  }
  int n  = col0 + (lane & 15);
  int mb = row0 + ((lane & 16) ? 8 : 0);
  float bv = bias ? bias[n] : 0.f;
#pragma unroll
  for (int r = 0; r < 8; ++r) {
    float v = acc[r] + bv;
    if (act == 1) v = fmaxf(v, 0.f);
    out[(size_t)(mb + r) * N + n] = v;
  }
}

// ---------------- init / elementwise ----------------
__global__ void k_fill_f32(float* p, float v, int n){
  int t = blockIdx.x * blockDim.x + threadIdx.x;
  if (t < n) p[t] = v;
}
__global__ void k_fill_u32(unsigned* p, unsigned v, int n){
  int t = blockIdx.x * blockDim.x + threadIdx.x;
  if (t < n) p[t] = v;
}
__global__ void k_mean(float* agg, const float* cnt, int n){
  int t = blockIdx.x * blockDim.x + threadIdx.x;
  if (t < n) agg[t] /= fmaxf(cnt[t >> 7], 1.f);
}
__global__ void k_scale(const float* __restrict__ xin, const float* __restrict__ sc,
                        const float* __restrict__ nm, float* __restrict__ xout, int n){
  int t = blockIdx.x * blockDim.x + threadIdx.x;
  if (t < n) {
    int i = t >> 7;
    xout[t] = (nm[i] > 0.5f) ? xin[t] * sc[i] : 0.f;
  }
}
__global__ void k_emf(float* emf, const float* nm, const int* src, const int* dst, int e_){
  int e = blockIdx.x * blockDim.x + threadIdx.x;
  if (e < e_) emf[e] *= nm[src[e]] * nm[dst[e]];
}

// ---------------- scatter (warp per edge) ----------------
__global__ void k_scatter(const float* __restrict__ x, const int* __restrict__ src,
                          const int* __restrict__ dst, const float* __restrict__ emf,
                          float* __restrict__ agg, float* cnt, int e_){
  int e = (blockIdx.x * blockDim.x + threadIdx.x) >> 5;
  if (e >= e_) return;
  float w = emf[e];
  if (w == 0.f) return;
  int lane = threadIdx.x & 31;
  int s = src[e], d = dst[e];
  float4 xv = ((const float4*)(x + (size_t)s * Fn))[lane];
  float* ap = agg + (size_t)d * Fn + lane * 4;
  atomicAdd(ap + 0, xv.x * w);
  atomicAdd(ap + 1, xv.y * w);
  atomicAdd(ap + 2, xv.z * w);
  atomicAdd(ap + 3, xv.w * w);
  if (cnt && lane == 0) atomicAdd(&cnt[d], w);
}

// ---------------- scores & pooling ----------------
__global__ void k_norm(const float* w, float* out){
  __shared__ float sm[128];
  float v = w[threadIdx.x];
  sm[threadIdx.x] = v * v;
  __syncthreads();
  for (int s = 64; s > 0; s >>= 1) {
    if ((int)threadIdx.x < s) sm[threadIdx.x] += sm[threadIdx.x + s];
    __syncthreads();
  }
  if (threadIdx.x == 0) out[0] = sqrtf(sm[0]);
}
__global__ void k_score1(const float* __restrict__ x, const float* __restrict__ w,
                         const float* nrm, float* sc){
  int i = (blockIdx.x * blockDim.x + threadIdx.x) >> 5;
  if (i >= NBn) return;
  int lane = threadIdx.x & 31;
  float4 xv = ((const float4*)(x + (size_t)i * Fn))[lane];
  float4 wv = ((const float4*)w)[lane];
  float p = xv.x * wv.x + xv.y * wv.y + xv.z * wv.z + xv.w * wv.w;
  p = wred32(p);
  if (lane == 0) sc[i] = tanhf(p / nrm[0]);
}
__global__ void k_score_gc(const float* __restrict__ agg, const float* __restrict__ x,
                           const float* __restrict__ wrel, const float* brel,
                           const float* __restrict__ wroot, float* sc){
  int i = (blockIdx.x * blockDim.x + threadIdx.x) >> 5;
  if (i >= NBn) return;
  int lane = threadIdx.x & 31;
  float4 av = ((const float4*)(agg + (size_t)i * Fn))[lane];
  float4 xv = ((const float4*)(x + (size_t)i * Fn))[lane];
  float4 wr = ((const float4*)wrel)[lane];
  float4 wo = ((const float4*)wroot)[lane];
  float p = av.x * wr.x + av.y * wr.y + av.z * wr.z + av.w * wr.w
          + xv.x * wo.x + xv.y * wo.y + xv.z * wo.z + xv.w * wo.w;
  p = wred32(p);
  if (lane == 0) sc[i] = tanhf(p + brel[0]);
}
// per-graph top-K by rank counting (stable: ties broken by lower index)
__global__ void k_pool(const float* __restrict__ sc, float* nmask, int K){
  __shared__ float s[Nn];
  int g = blockIdx.x, j = threadIdx.x, i = g * Nn + j;
  float v = (nmask[i] > 0.5f) ? sc[i] : -__builtin_inff();
  s[j] = v;
  __syncthreads();
  int rank = 0;
  for (int t = 0; t < Nn; ++t) {
    float o = s[t];
    rank += (o > v) || (o == v && t < j);
  }
  __syncthreads();
  nmask[i] = (rank < K) ? 1.f : 0.f;
}
__global__ void k_gap(const float* __restrict__ x, float* h, float invK){
  int g = blockIdx.x, c = threadIdx.x;
  const float* p = x + (size_t)g * Nn * Fn + c;
  float s = 0.f;
  for (int j = 0; j < Nn; ++j) s += p[(size_t)j * Fn];
  h[g * Fn + c] += s * invK;
}

// ---------------- GAT ----------------
__global__ void k_alsd(const float* __restrict__ xs, const float* __restrict__ a_s,
                       const float* __restrict__ a_d, float* als, float* ald){
  int i = (blockIdx.x * blockDim.x + threadIdx.x) >> 5;
  if (i >= NBn) return;
  int lane = threadIdx.x & 31, head = lane >> 3;
  float4 xv  = ((const float4*)(xs + (size_t)i * Fn))[lane];
  float4 sv  = ((const float4*)a_s)[lane];
  float4 dv  = ((const float4*)a_d)[lane];
  float ps = xv.x * sv.x + xv.y * sv.y + xv.z * sv.z + xv.w * sv.w;
  float pd = xv.x * dv.x + xv.y * dv.y + xv.z * dv.z + xv.w * dv.w;
  ps = hred8(ps);  pd = hred8(pd);
  if ((lane & 7) == 0) { als[i * 4 + head] = ps; ald[i * 4 + head] = pd; }
}
// v[d][h] = sum_c w_e[d, h*32+c] * a_e[h, c]   (7x4)
__global__ void k_v(const float* __restrict__ w_e, const float* __restrict__ a_e, float* v){
  int t = threadIdx.x;
  if (t < 28) {
    int d = t >> 2, h = t & 3;
    float s = 0.f;
    for (int c = 0; c < 32; ++c) s += w_e[d * 128 + h * 32 + c] * a_e[h * 32 + c];
    v[t] = s;
  }
}
__global__ void k_ea_reduce(const float* __restrict__ ea, const float* __restrict__ emf,
                            float* sums, int e_){
  __shared__ float sm[8];
  if (threadIdx.x < 8) sm[threadIdx.x] = 0.f;
  __syncthreads();
  int e = blockIdx.x * blockDim.x + threadIdx.x;
  if (e < e_ && emf[e] > 0.f) {
#pragma unroll
    for (int d = 0; d < 7; ++d) atomicAdd(&sm[d], ea[(size_t)e * 7 + d]);
    atomicAdd(&sm[7], 1.f);
  }
  __syncthreads();
  if (threadIdx.x < 8) atomicAdd(&sums[threadIdx.x], sm[threadIdx.x]);
}
__global__ void k_alse(const float* sums, const float* v, float* alse){
  int h = threadIdx.x;
  if (h < 4) {
    float inv = 1.f / fmaxf(sums[7], 1.f);
    float s = 0.f;
    for (int d = 0; d < 7; ++d) s += sums[d] * inv * v[d * 4 + h];
    alse[h] = s;
  }
}
__global__ void k_alpha(const float* __restrict__ ea, const float* __restrict__ emf,
                        const int* __restrict__ src, const int* __restrict__ dst,
                        const float* __restrict__ als, const float* __restrict__ ald,
                        const float* __restrict__ v, float* alpha, unsigned* mo, int e4){
  int t = blockIdx.x * blockDim.x + threadIdx.x;
  if (t >= e4) return;
  int e = t >> 2, h = t & 3;
  if (emf[e] == 0.f) { alpha[t] = -__builtin_inff(); return; }
  float ae = 0.f;
#pragma unroll
  for (int d = 0; d < 7; ++d) ae += ea[(size_t)e * 7 + d] * v[d * 4 + h];
  float a = als[src[e] * 4 + h] + ald[dst[e] * 4 + h] + ae;
  a = (a > 0.f) ? a : 0.2f * a;
  alpha[t] = a;
  atomicMax(&mo[dst[e] * 4 + h], f2o(a));
}
__global__ void k_self(const float* __restrict__ als, const float* __restrict__ ald,
                       const float* alse, const float* __restrict__ nm,
                       float* aself, unsigned* mo, int n4){
  int t = blockIdx.x * blockDim.x + threadIdx.x;
  if (t >= n4) return;
  int i = t >> 2, h = t & 3;
  if (nm[i] <= 0.5f) { aself[t] = -__builtin_inff(); return; }
  float a = als[t] + ald[t] + alse[h];
  a = (a > 0.f) ? a : 0.2f * a;
  aself[t] = a;
  atomicMax(&mo[t], f2o(a));
}
__global__ void k_mfin(unsigned* mo, int n4){
  int t = blockIdx.x * blockDim.x + threadIdx.x;
  if (t >= n4) return;
  float f = o2f(mo[t]);
  if (__builtin_isinf(f)) f = 0.f;
  ((float*)mo)[t] = f;
}
__global__ void k_ex(const float* __restrict__ alpha, const float* __restrict__ mf,
                     const float* __restrict__ emf, const int* __restrict__ src,
                     const int* __restrict__ dst, const float* __restrict__ xs,
                     float* den, float* num, int e_){
  int e = (blockIdx.x * blockDim.x + threadIdx.x) >> 5;
  if (e >= e_) return;
  if (emf[e] == 0.f) return;
  int lane = threadIdx.x & 31, head = lane >> 3;
  int s = src[e], d = dst[e];
  float ex = expf(alpha[e * 4 + head] - mf[d * 4 + head]);
  if ((lane & 7) == 0) atomicAdd(&den[d * 4 + head], ex);
  float4 xv = ((const float4*)(xs + (size_t)s * Fn))[lane];
  float* np = num + (size_t)d * Fn + lane * 4;
  atomicAdd(np + 0, ex * xv.x);
  atomicAdd(np + 1, ex * xv.y);
  atomicAdd(np + 2, ex * xv.z);
  atomicAdd(np + 3, ex * xv.w);
}
__global__ void k_gatfin(const float* __restrict__ num, const float* __restrict__ xs,
                         const float* __restrict__ aself, const float* __restrict__ mf,
                         const float* __restrict__ den, const float* __restrict__ nm,
                         const float* __restrict__ g_b, float* __restrict__ out){
  int i = (blockIdx.x * blockDim.x + threadIdx.x) >> 5;
  if (i >= NBn) return;
  int lane = threadIdx.x & 31, head = lane >> 3;
  float exs = 0.f;
  if (nm[i] > 0.5f) exs = expf(aself[i * 4 + head] - mf[i * 4 + head]);
  float dent = fmaxf(den[i * 4 + head] + exs, 1e-16f);
  float4 nv = ((const float4*)(num + (size_t)i * Fn))[lane];
  float4 xv = ((const float4*)(xs + (size_t)i * Fn))[lane];
  float4 bv = ((const float4*)g_b)[lane];
  float4 o;
  o.x = fmaxf((nv.x + exs * xv.x) / dent + bv.x, 0.f);
  o.y = fmaxf((nv.y + exs * xv.y) / dent + bv.y, 0.f);
  o.z = fmaxf((nv.z + exs * xv.z) / dent + bv.z, 0.f);
  o.w = fmaxf((nv.w + exs * xv.w) / dent + bv.w, 0.f);
  ((float4*)(out + (size_t)i * Fn))[lane] = o;
}

// ---------------- head final layer ----------------
__global__ void k_l3(const float* __restrict__ h2, const float* __restrict__ w,
                     const float* b, float* out){
  int g = (blockIdx.x * blockDim.x + threadIdx.x) >> 5;
  if (g >= Bn) return;
  int lane = threadIdx.x & 31, c = lane * 2;
  float p = h2[g * 64 + c] * w[c] + h2[g * 64 + c + 1] * w[c + 1];
  p = wred32(p);
  if (lane == 0) out[g] = 1.f / (1.f + expf(-(p + b[0])));
}

extern "C" void kernel_launch(void* const* d_in, const int* in_sizes, int n_in,
                              void* d_out, int out_size, void* d_ws, size_t ws_size,
                              hipStream_t stream) {
  (void)in_sizes; (void)n_in; (void)out_size; (void)ws_size;
  const float* x    = (const float*)d_in[0];
  const int*   ei   = (const int*)d_in[1];
  const float* ea   = (const float*)d_in[2];
  const float* c1wl = (const float*)d_in[3];
  const float* c1bl = (const float*)d_in[4];
  const float* c1wr = (const float*)d_in[5];
  const float* p1w  = (const float*)d_in[6];
  const float* c2wl = (const float*)d_in[7];
  const float* c2bl = (const float*)d_in[8];
  const float* c2wr = (const float*)d_in[9];
  const float* p2wr = (const float*)d_in[10];
  const float* p2br = (const float*)d_in[11];
  const float* p2wo = (const float*)d_in[12];
  const float* gw   = (const float*)d_in[13];
  const float* gas  = (const float*)d_in[14];
  const float* gad  = (const float*)d_in[15];
  const float* gwe  = (const float*)d_in[16];
  const float* gae  = (const float*)d_in[17];
  const float* gb   = (const float*)d_in[18];
  const float* p3wr = (const float*)d_in[19];
  const float* p3br = (const float*)d_in[20];
  const float* p3wo = (const float*)d_in[21];
  const float* l1w  = (const float*)d_in[22];
  const float* l1b  = (const float*)d_in[23];
  const float* l2w  = (const float*)d_in[24];
  const float* l2b  = (const float*)d_in[25];
  const float* l3w  = (const float*)d_in[26];
  const float* l3b  = (const float*)d_in[27];
  const int* src = ei;
  const int* dst = ei + En;

  // workspace layout (floats)
  float* w = (float*)d_ws;
  size_t off = 0;
  float* XA   = w + off; off += (size_t)NBn * Fn;
  float* XB   = w + off; off += (size_t)NBn * Fn;
  float* AG   = w + off; off += (size_t)NBn * Fn;   // agg / mean / GAT num
  float* XS   = w + off; off += (size_t)NBn * Fn;   // GAT xs
  float* CNT  = w + off; off += NBn;
  float* EMF  = w + off; off += En;
  float* NM   = w + off; off += NBn;
  float* SC   = w + off; off += NBn;
  float* ALS  = w + off; off += (size_t)NBn * 4;
  float* ALD  = w + off; off += (size_t)NBn * 4;
  float* MO   = w + off; off += (size_t)NBn * 4;    // ordered-u32 max, then float m
  float* ASF  = w + off; off += (size_t)NBn * 4;
  float* DEN  = w + off; off += (size_t)NBn * 4;
  float* ALPH = w + off; off += (size_t)En * 4;
  float* Hb   = w + off; off += (size_t)Bn * Fn;
  float* H1   = w + off; off += (size_t)Bn * Fn;
  float* H2   = w + off; off += (size_t)Bn * 64;
  float* SMALL= w + off; off += 64;                 // [0]=norm, [8..15]=ea sums, [16..43]=v, [48..51]=alse
  // packed bf16 weights (fragment-major)
  __bf16* PW = (__bf16*)(w + off);
  __bf16* P1L = PW;              // 128x128 -> 16384 bf16
  __bf16* P1R = PW + 16384;
  __bf16* P2L = PW + 16384 * 2;
  __bf16* P2R = PW + 16384 * 3;
  __bf16* PGW = PW + 16384 * 4;
  __bf16* PL1 = PW + 16384 * 5;
  __bf16* PL2 = PW + 16384 * 6;  // 128x64 -> 8192 bf16

  const size_t bigBytes = (size_t)NBn * Fn * sizeof(float);
  const int TB = 256;
  const int gElem = (NBn * Fn) / TB;     // 32768
  const int gNodeW = (NBn * 32) / TB;    // 8192
  const int gEdgeW = (En * 32) / TB;     // 131072
  const int gE   = En / TB;              // 4096
  const int gE4  = (En * 4) / TB;        // 16384
  const int gN4  = (NBn * 4) / TB;       // 1024

  // ---- pack all weights into WMMA-fragment bf16 layout (L2-resident) ----
  k_packw<<<32, TB, 0, stream>>>(c1wl, P1L, 128);
  k_packw<<<32, TB, 0, stream>>>(c1wr, P1R, 128);
  k_packw<<<32, TB, 0, stream>>>(c2wl, P2L, 128);
  k_packw<<<32, TB, 0, stream>>>(c2wr, P2R, 128);
  k_packw<<<32, TB, 0, stream>>>(gw,   PGW, 128);
  k_packw<<<32, TB, 0, stream>>>(l1w,  PL1, 128);
  k_packw<<<16, TB, 0, stream>>>(l2w,  PL2, 64);

  // ---- init ----
  k_fill_f32<<<gE, TB, 0, stream>>>(EMF, 1.f, En);
  k_fill_f32<<<NBn / TB, TB, 0, stream>>>(NM, 1.f, NBn);
  hipMemsetAsync(Hb, 0, (size_t)Bn * Fn * sizeof(float), stream);

  // ---- block 1: SAGE -> TopK(0.8) -> gap ----
  hipMemsetAsync(AG, 0, bigBytes, stream);
  hipMemsetAsync(CNT, 0, NBn * sizeof(float), stream);
  k_scatter<<<gEdgeW, TB, 0, stream>>>(x, src, dst, EMF, AG, CNT, En);
  k_mean<<<gElem, TB, 0, stream>>>(AG, CNT, NBn * Fn);
  k_gemm<true><<<NBn / 16, 256, 0, stream>>>(AG, P1L, x, P1R, c1bl, XB, 128, 1);
  k_norm<<<1, 128, 0, stream>>>(p1w, SMALL);
  k_score1<<<gNodeW, TB, 0, stream>>>(XB, p1w, SMALL, SC);
  k_pool<<<Bn, Nn, 0, stream>>>(SC, NM, K1n);
  k_scale<<<gElem, TB, 0, stream>>>(XB, SC, NM, XA, NBn * Fn);
  k_emf<<<gE, TB, 0, stream>>>(EMF, NM, src, dst, En);
  k_gap<<<Bn, Fn, 0, stream>>>(XA, Hb, 1.f / K1n);

  // ---- block 2: SAGE -> SAGPool(0.5) -> gap ----
  hipMemsetAsync(AG, 0, bigBytes, stream);
  hipMemsetAsync(CNT, 0, NBn * sizeof(float), stream);
  k_scatter<<<gEdgeW, TB, 0, stream>>>(XA, src, dst, EMF, AG, CNT, En);
  k_mean<<<gElem, TB, 0, stream>>>(AG, CNT, NBn * Fn);
  k_gemm<true><<<NBn / 16, 256, 0, stream>>>(AG, P2L, XA, P2R, c2bl, XB, 128, 1);
  hipMemsetAsync(AG, 0, bigBytes, stream);
  k_scatter<<<gEdgeW, TB, 0, stream>>>(XB, src, dst, EMF, AG, nullptr, En);
  k_score_gc<<<gNodeW, TB, 0, stream>>>(AG, XB, p2wr, p2br, p2wo, SC);
  k_pool<<<Bn, Nn, 0, stream>>>(SC, NM, K2n);
  k_scale<<<gElem, TB, 0, stream>>>(XB, SC, NM, XA, NBn * Fn);
  k_emf<<<gE, TB, 0, stream>>>(EMF, NM, src, dst, En);
  k_gap<<<Bn, Fn, 0, stream>>>(XA, Hb, 1.f / K2n);

  // ---- block 3: GAT -> SAGPool(0.5) -> gap ----
  k_gemm<false><<<NBn / 16, 256, 0, stream>>>(XA, PGW, nullptr, nullptr, nullptr, XS, 128, 0);
  k_alsd<<<gNodeW, TB, 0, stream>>>(XS, gas, gad, ALS, ALD);
  k_v<<<1, 32, 0, stream>>>(gwe, gae, SMALL + 16);
  hipMemsetAsync(SMALL + 8, 0, 8 * sizeof(float), stream);
  k_ea_reduce<<<gE, TB, 0, stream>>>(ea, EMF, SMALL + 8, En);
  k_alse<<<1, 4, 0, stream>>>(SMALL + 8, SMALL + 16, SMALL + 48);
  k_fill_u32<<<gN4, TB, 0, stream>>>((unsigned*)MO, 0x007FFFFFu, NBn * 4);  // f2o(-inf)
  k_alpha<<<gE4, TB, 0, stream>>>(ea, EMF, src, dst, ALS, ALD, SMALL + 16, ALPH, (unsigned*)MO, En * 4);
  k_self<<<gN4, TB, 0, stream>>>(ALS, ALD, SMALL + 48, NM, ASF, (unsigned*)MO, NBn * 4);
  k_mfin<<<gN4, TB, 0, stream>>>((unsigned*)MO, NBn * 4);
  hipMemsetAsync(DEN, 0, (size_t)NBn * 4 * sizeof(float), stream);
  hipMemsetAsync(AG, 0, bigBytes, stream);
  k_ex<<<gEdgeW, TB, 0, stream>>>(ALPH, MO, EMF, src, dst, XS, DEN, AG, En);
  k_gatfin<<<gNodeW, TB, 0, stream>>>(AG, XS, ASF, MO, DEN, NM, gb, XB);
  hipMemsetAsync(AG, 0, bigBytes, stream);
  k_scatter<<<gEdgeW, TB, 0, stream>>>(XB, src, dst, EMF, AG, nullptr, En);
  k_score_gc<<<gNodeW, TB, 0, stream>>>(AG, XB, p3wr, p3br, p3wo, SC);
  k_pool<<<Bn, Nn, 0, stream>>>(SC, NM, K3n);
  k_scale<<<gElem, TB, 0, stream>>>(XB, SC, NM, XA, NBn * Fn);
  k_gap<<<Bn, Fn, 0, stream>>>(XA, Hb, 1.f / K3n);

  // ---- MLP head ----
  k_gemm<false><<<Bn / 16, 256, 0, stream>>>(Hb, PL1, nullptr, nullptr, l1b, H1, 128, 1);
  k_gemm<false><<<Bn / 16, 128, 0, stream>>>(H1, PL2, nullptr, nullptr, l2b, H2, 64, 1);
  k_l3<<<(Bn * 32) / TB, TB, 0, stream>>>(H2, l3w, l3b, (float*)d_out);
}
